// CausalWanSelfAttention_90340342104723
// MI455X (gfx1250) — compile-verified, matrix operand order checked by reading, not run on hardware
//
#include <hip/hip_runtime.h>
#include <cstddef>

// ---------------- problem constants ----------------
#define LQ     1560
#define NH     12
#define HD     128
#define CACHE  9360
#define MAXATT 9360
#define KT     64          // key-tile per iteration
#define PADK   9408        // CACHE rounded up to multiple of KT (147*64)

// 1/sqrt(128) * log2(e): folded into Q so softmax can run in exp2 domain
#define QSCALE 0.12751744564390826f

// device exp2 via AMDGPU builtin -> v_exp_f32
#define EXP2F(x) __builtin_amdgcn_exp2f(x)

typedef __attribute__((ext_vector_type(16))) __bf16 v16bf;
typedef __attribute__((ext_vector_type(8)))  __bf16 v8bf;
typedef __attribute__((ext_vector_type(8)))  float  v8f;

// workspace layout (bytes)
#define QB_BYTES ((size_t)NH * LQ * HD * 2)
#define KW_BYTES ((size_t)NH * PADK * HD * 2)
#define VT_BYTES ((size_t)NH * HD * PADK * 2)

// Build a 16-element bf16 WMMA fragment from two contiguous 16B groups.
static __device__ __forceinline__ v16bf frag16(const __bf16* p0, const __bf16* p1) {
  v8bf a = *(const v8bf*)p0;
  v8bf b = *(const v8bf*)p1;
  v16bf r;
#pragma unroll
  for (int e = 0; e < 8; ++e) { r[e] = a[e]; r[e + 8] = b[e]; }
  return r;
}

// ---------------- prep kernel 1: RoPE(Q)*QSCALE -> bf16 Qb[h][q][d] ----------------
__global__ void k_rope_q(const float* __restrict__ q, const float* __restrict__ cosb,
                         const float* __restrict__ sinb, __bf16* __restrict__ Qb) {
  int i = blockIdx.x * blockDim.x + threadIdx.x;
  if (i >= NH * LQ * (HD / 2)) return;
  int d2 = i & 63;
  int hh = (i >> 6) % NH;
  int qi = i / (64 * NH);
  float c = cosb[qi * 64 + d2], s = sinb[qi * 64 + d2];
  int ib = ((qi * NH + hh) * HD) + 2 * d2;
  float x1 = q[ib], x2 = q[ib + 1];
  __bf16* o = Qb + ((size_t)(hh * LQ + qi) * HD) + 2 * d2;
  o[0] = (__bf16)((x1 * c - x2 * s) * QSCALE);
  o[1] = (__bf16)((x2 * c + x1 * s) * QSCALE);
}

// ------- prep kernel 2: window K = [cache_k ; RoPE(k)] -> bf16 Kw[h][kpos][d] -------
__global__ void k_build_k(const float* __restrict__ k, const float* __restrict__ cache_k,
                          const float* __restrict__ cosb, const float* __restrict__ sinb,
                          const int* __restrict__ leip, __bf16* __restrict__ Kw) {
  int i = blockIdx.x * blockDim.x + threadIdx.x;
  if (i >= NH * PADK * (HD / 2)) return;
  int d2 = i & 63;
  int hh = (i >> 6) % NH;
  int kp = i / (64 * NH);
  int lei = *leip;
  int win_start = (lei > MAXATT) ? (lei - MAXATT) : 0;
  int Lk = lei - win_start;
  int local_start = lei - LQ;
  __bf16* o = Kw + ((size_t)(hh * PADK + kp) * HD) + 2 * d2;
  if (kp >= Lk) { o[0] = (__bf16)0.f; o[1] = (__bf16)0.f; return; }
  int ci = win_start + kp;
  float r0, r1;
  if (ci < local_start) {
    int ib = ((ci * NH + hh) * HD) + 2 * d2;
    r0 = cache_k[ib]; r1 = cache_k[ib + 1];
  } else {
    int l = ci - local_start;
    int ib = ((l * NH + hh) * HD) + 2 * d2;
    float x1 = k[ib], x2 = k[ib + 1];
    float c = cosb[l * 64 + d2], s = sinb[l * 64 + d2];
    r0 = x1 * c - x2 * s;
    r1 = x2 * c + x1 * s;
  }
  o[0] = (__bf16)r0; o[1] = (__bf16)r1;
}

// ---- prep kernel 3: window V transposed -> bf16 Vt[h][d][kpos] (LDS tile transpose) ----
__global__ void k_build_vt(const float* __restrict__ v, const float* __restrict__ cache_v,
                           const int* __restrict__ leip, __bf16* __restrict__ Vt) {
  __shared__ __bf16 tile[32][33];
  int hh = blockIdx.z;
  int d0 = blockIdx.y * 32;
  int kp0 = blockIdx.x * 32;
  int lei = *leip;
  int win_start = (lei > MAXATT) ? (lei - MAXATT) : 0;
  int Lk = lei - win_start;
  int local_start = lei - LQ;
#pragma unroll
  for (int yy = threadIdx.y; yy < 32; yy += 8) {
    int kp = kp0 + yy;
    int d = d0 + threadIdx.x;
    float val = 0.f;
    if (kp < Lk) {
      int ci = win_start + kp;
      if (ci < local_start) val = cache_v[((size_t)(ci * NH + hh) * HD) + d];
      else                  val = v[((size_t)((ci - local_start) * NH + hh) * HD) + d];
    }
    tile[yy][threadIdx.x] = (__bf16)val;
  }
  __syncthreads();
#pragma unroll
  for (int yy = threadIdx.y; yy < 32; yy += 8) {
    int d = d0 + yy;
    int kp = kp0 + threadIdx.x;
    Vt[((size_t)(hh * HD + d) * PADK) + kp] = tile[threadIdx.x][yy];
  }
}

// ---------------- flash-attention kernel ----------------
// block = 128 threads = 4 waves; wave w owns query rows q0 + 16w .. +15.
// grid = (ceil(LQ/64), NH)
__launch_bounds__(128)
__global__ void k_attn(const __bf16* __restrict__ Qb, const __bf16* __restrict__ Kw,
                       const __bf16* __restrict__ Vt, const int* __restrict__ leip,
                       float* __restrict__ out) {
  __shared__ __align__(16) __bf16 sK[KT * HD];      // [kpos][d]   16 KB
  __shared__ __align__(16) __bf16 sV[HD * KT];      // [d][kpos]   16 KB
  __shared__ __align__(16) __bf16 sP[4][16 * KT];   // per-wave P   8 KB

  const int head = blockIdx.y;
  const int q0 = blockIdx.x * 64;
  const int tid = threadIdx.x;
  const int wave = tid >> 5;
  const int lane = tid & 31;
  const int n16 = lane & 15;
  const int hh = lane >> 4;

  const int lei = *leip;
  const int win_start = (lei > MAXATT) ? (lei - MAXATT) : 0;
  const int Lk = lei - win_start;
  const int T = (Lk + KT - 1) / KT;

  // ---- load Q A-fragments (16x128 per wave, 4 chunks of K-dim 32) ----
  int qrow = q0 + wave * 16 + n16;
  if (qrow >= LQ) qrow = LQ - 1;
  const __bf16* qp = Qb + (size_t)(head * LQ + qrow) * HD;
  v16bf qf[4];
#pragma unroll
  for (int c = 0; c < 4; ++c) {
    const __bf16* p = qp + 32 * c + 8 * hh;
    qf[c] = frag16(p, p + 16);
  }

  v8f of[8];
#pragma unroll
  for (int nb = 0; nb < 8; ++nb) { v8f z = {}; of[nb] = z; }
  float mi[8], li[8];
#pragma unroll
  for (int r = 0; r < 8; ++r) { mi[r] = -1e30f; li[r] = 0.f; }

  const __bf16* Kbase = Kw + (size_t)head * PADK * HD;
  const __bf16* Vbase = Vt + (size_t)head * HD * PADK;
  __bf16* sPw = sP[wave];

  // staging: each thread moves 128B of K and 128B of V per tile
  float4 kr[8], vr[8];
  {
    const float4* kg = (const float4*)(Kbase) + tid * 8;
    const float4* vg = (const float4*)(Vbase + (size_t)tid * PADK);
#pragma unroll
    for (int j = 0; j < 8; ++j) { kr[j] = kg[j]; vr[j] = vg[j]; }
  }

  for (int t = 0; t < T; ++t) {
    const int kt0 = t * KT;
    __syncthreads();
    // commit staged tile to LDS
    {
      float4* kd = (float4*)sK + tid * 8;
      float4* vd = (float4*)(sV + tid * KT);
#pragma unroll
      for (int j = 0; j < 8; ++j) { kd[j] = kr[j]; vd[j] = vr[j]; }
    }
    // issue global loads for next tile (overlaps with compute below)
    if (t + 1 < T) {
      const int n0 = kt0 + KT;
      const float4* kg = (const float4*)(Kbase + (size_t)n0 * HD) + tid * 8;
      const float4* vg = (const float4*)(Vbase + (size_t)tid * PADK + n0);
#pragma unroll
      for (int j = 0; j < 8; ++j) { kr[j] = kg[j]; vr[j] = vg[j]; }
    }
    if (t + 2 < T) {
      const int p0 = kt0 + 2 * KT;
      __builtin_prefetch(Kbase + (size_t)p0 * HD + tid * 64, 0, 0);
      __builtin_prefetch(Vbase + (size_t)tid * PADK + p0, 0, 0);
    }
    __syncthreads();

    // ---- GEMM1: S(16x64) = Q(16x128) * K^T, 4 N-fragments x 4 K-chunks ----
    // i = c*4 + nbi ; pipelined 2-deep so ds_loads overlap WMMA.
    v8f s[4];
#pragma unroll
    for (int j = 0; j < 4; ++j) { v8f z = {}; s[j] = z; }
    {
      auto loadK1 = [&](int i) -> v16bf {
        const int c = i >> 2, nbi = i & 3;
        const __bf16* p = sK + (nbi * 16 + n16) * HD + 32 * c + 8 * hh;
        return frag16(p, p + 16);
      };
      v16bf bb[2];
      bb[0] = loadK1(0);
      bb[1] = loadK1(1);
#pragma unroll
      for (int i = 0; i < 16; ++i) {
        v16bf cur = bb[i & 1];
        if (i + 2 < 16) bb[i & 1] = loadK1(i + 2);
        s[i & 3] = __builtin_amdgcn_wmma_f32_16x16x32_bf16(false, qf[i >> 2], false, cur,
                                                           (short)0, s[i & 3], false, false);
      }
    }

    // ---- online softmax (exp2 domain; scale pre-folded into Q) ----
    float a[4][8];
#pragma unroll
    for (int nbi = 0; nbi < 4; ++nbi) {
      const bool msk = (kt0 + nbi * 16 + n16) >= Lk;
#pragma unroll
      for (int r = 0; r < 8; ++r) a[nbi][r] = msk ? -1e30f : s[nbi][r];
    }
#pragma unroll
    for (int r = 0; r < 8; ++r) {
      float mx = fmaxf(fmaxf(a[0][r], a[1][r]), fmaxf(a[2][r], a[3][r]));
      mx = fmaxf(mx, __shfl_xor(mx, 1));
      mx = fmaxf(mx, __shfl_xor(mx, 2));
      mx = fmaxf(mx, __shfl_xor(mx, 4));
      mx = fmaxf(mx, __shfl_xor(mx, 8));
      const float mn = fmaxf(mi[r], mx);
      const float corr = EXP2F(mi[r] - mn);
      mi[r] = mn;
      const float e0 = EXP2F(a[0][r] - mn);
      const float e1 = EXP2F(a[1][r] - mn);
      const float e2 = EXP2F(a[2][r] - mn);
      const float e3 = EXP2F(a[3][r] - mn);
      float rs = (e0 + e1) + (e2 + e3);
      rs += __shfl_xor(rs, 1);
      rs += __shfl_xor(rs, 2);
      rs += __shfl_xor(rs, 4);
      rs += __shfl_xor(rs, 8);
      li[r] = li[r] * corr + rs;
#pragma unroll
      for (int nb = 0; nb < 8; ++nb) of[nb][r] *= corr;
      // stash P (C-layout -> LDS row-major 16x64, wave-private)
      __bf16* pr = sPw + (r + 8 * hh) * KT + n16;
      pr[0]  = (__bf16)e0;
      pr[16] = (__bf16)e1;
      pr[32] = (__bf16)e2;
      pr[48] = (__bf16)e3;
    }

    // ---- GEMM2: O(16x128) += P(16x64) * V(64x128) ----
    v16bf pa[2];
#pragma unroll
    for (int ac = 0; ac < 2; ++ac) {
      const __bf16* p = sPw + n16 * KT + 32 * ac + 8 * hh;
      pa[ac] = frag16(p, p + 16);
    }
    {
      auto loadV1 = [&](int i) -> v16bf {
        const int ac = i >> 3, nb = i & 7;
        const __bf16* p = sV + (nb * 16 + n16) * KT + 32 * ac + 8 * hh;
        return frag16(p, p + 16);
      };
      v16bf bb[2];
      bb[0] = loadV1(0);
      bb[1] = loadV1(1);
#pragma unroll
      for (int i = 0; i < 16; ++i) {
        v16bf cur = bb[i & 1];
        if (i + 2 < 16) bb[i & 1] = loadV1(i + 2);
        of[i & 7] = __builtin_amdgcn_wmma_f32_16x16x32_bf16(false, pa[i >> 3], false, cur,
                                                            (short)0, of[i & 7], false, false);
      }
    }
  }

  // ---- epilogue: normalize and store ----
  float inv[8];
#pragma unroll
  for (int r = 0; r < 8; ++r) inv[r] = 1.0f / li[r];
#pragma unroll
  for (int nb = 0; nb < 8; ++nb) {
#pragma unroll
    for (int r = 0; r < 8; ++r) {
      const int qq = q0 + wave * 16 + r + 8 * hh;
      if (qq < LQ)
        out[((size_t)qq * NH + head) * HD + nb * 16 + n16] = of[nb][r] * inv[r];
    }
  }
}

// ---------------- host launcher ----------------
extern "C" void kernel_launch(void* const* d_in, const int* in_sizes, int n_in,
                              void* d_out, int out_size, void* d_ws, size_t ws_size,
                              hipStream_t stream) {
  const float* q       = (const float*)d_in[0];
  const float* k       = (const float*)d_in[1];
  const float* v       = (const float*)d_in[2];
  const float* cosb    = (const float*)d_in[3];
  const float* sinb    = (const float*)d_in[4];
  const float* cache_k = (const float*)d_in[5];
  const float* cache_v = (const float*)d_in[6];
  const int*   lei     = (const int*)d_in[7];
  float* out = (float*)d_out;

  char* ws = (char*)d_ws;
  __bf16* Qb = (__bf16*)(ws);
  __bf16* Kw = (__bf16*)(ws + QB_BYTES);
  __bf16* Vt = (__bf16*)(ws + QB_BYTES + KW_BYTES);

  {
    int n = NH * LQ * (HD / 2);
    k_rope_q<<<(n + 255) / 256, 256, 0, stream>>>(q, cosb, sinb, Qb);
  }
  {
    int n = NH * PADK * (HD / 2);
    k_build_k<<<(n + 255) / 256, 256, 0, stream>>>(k, cache_k, cosb, sinb, lei, Kw);
  }
  {
    dim3 g(PADK / 32, HD / 32, NH);
    k_build_vt<<<g, dim3(32, 8), 0, stream>>>(v, cache_v, lei, Vt);
  }
  {
    dim3 g((LQ + 63) / 64, NH);
    k_attn<<<g, 128, 0, stream>>>(Qb, Kw, Vt, lei, out);
  }
}
